// regression_Transformer_32263794328446
// MI455X (gfx1250) — matrix operand order repaired and checked by reading, not compiled, not run on hardware
//
#include <hip/hip_runtime.h>
#include <hip/hip_bf16.h>

// ---------------------------------------------------------------------------
// Transformer forward for MI455X (gfx1250): bf16 WMMA GEMMs + flash attention.
// B=16, S=512, IN=32, D=768, H=12, DH=64, DFF=3072, L=6
// Weights pre-converted once per launch into bf16 fragment order; GEMM tiles
// are DMA'd into LDS by the Tensor Data Mover (double-buffered, TENSORcnt).
// ---------------------------------------------------------------------------

#define BB 16
#define SS 512
#define DIN 32
#define DD 768
#define HH_ 12
#define DH_ 64
#define DFF_ 3072
#define LL 6
#define MTOT (BB * SS)   // 8192 rows

typedef __bf16 v16bf __attribute__((ext_vector_type(16)));
typedef float  v8f   __attribute__((ext_vector_type(8)));

union Frag {
    v16bf        bf;
    unsigned int u[8];
};

__device__ __forceinline__ unsigned short f2bf(float f) {
    unsigned int u = __float_as_uint(f);
    u += 0x7FFFu + ((u >> 16) & 1u);   // round-to-nearest-even
    return (unsigned short)(u >> 16);
}

// ---------------------------------------------------------------------------
// Tensor Data Mover: 2D tile DMA global -> LDS (data_size = 8 bytes).
// D# layout per CDNA5 ISA ch.8 (group0: count/lds/global/type, group1: dims).
// This toolchain exposes the 6-arg builtin:
//   (uint32x4 g0, int32x8 g1, int32x4 g2, int32x4 g3, int32x8 pad, i32 cpol)
// ---------------------------------------------------------------------------
#if __has_builtin(__builtin_amdgcn_tensor_load_to_lds)
#define HAVE_TDM 1
typedef unsigned int v4u  __attribute__((ext_vector_type(4)));
typedef int          v8i_ __attribute__((ext_vector_type(8)));
typedef int          v4i_ __attribute__((ext_vector_type(4)));

__device__ __forceinline__ void tdm_load_2d(unsigned lds_off, const void* gptr,
                                            unsigned tile_d0, unsigned tile_d1,
                                            unsigned long long stride0_units)
{
    const unsigned long long ga = (unsigned long long)gptr;
    v4u g0;
    g0[0] = 1u;                                        // count=1 (user D#)
    g0[1] = lds_off;                                   // LDS byte address
    g0[2] = (unsigned)(ga & 0xFFFFFFFFu);              // global addr [31:0]
    g0[3] = (unsigned)((ga >> 32) & 0x01FFFFFFu)       // global addr [56:32]
            | (2u << 30);                              // type = 2 ("image")
    const unsigned td0 = 1u << 30, td1 = 1u << 30;     // huge dims: no OOB clip
    v8i_ g1;
    g1[0] = (int)(3u << 16);                           // data_size = 8 bytes
    g1[1] = (int)((td0 & 0xFFFFu) << 16);              // tensor_dim0 lo
    g1[2] = (int)((td0 >> 16) | ((td1 & 0xFFFFu) << 16));
    g1[3] = (int)((td1 >> 16) | (tile_d0 << 16));      // tile_dim0
    g1[4] = (int)tile_d1;                              // tile_dim1 (dim2 = 0)
    g1[5] = (int)(unsigned)(stride0_units & 0xFFFFFFFFull);
    g1[6] = (int)(unsigned)(stride0_units >> 32);      // stride1 = 0
    g1[7] = 0;
    const v4i_ z4 = {0, 0, 0, 0};
    const v8i_ z8 = {0, 0, 0, 0, 0, 0, 0, 0};
    __builtin_amdgcn_tensor_load_to_lds(g0, g1, z4, z4, z8, 0);
}
#else
#define HAVE_TDM 0
#endif

// ---------------------------------------------------------------------------
// Weight pre-transpose + fp32->bf16:  W[K,N] -> Wt[nb][kc][n(128)][k(32)]
// ---------------------------------------------------------------------------
__global__ __launch_bounds__(256) void w_pretr_kernel(
    const float* __restrict__ W, unsigned short* __restrict__ Wt,
    int K, int N)
{
    const size_t lmat = blockIdx.y;
    const float*          src = W  + lmat * (size_t)K * N;
    unsigned short*       dst = Wt + lmat * (size_t)K * N;
    const size_t base = ((size_t)blockIdx.x * 256 + threadIdx.x) * 8;
    const int    kk0  = (int)(base & 31);
    const int    nl   = (int)((base >> 5) & 127);
    const size_t rest = base >> 12;          // /(32*128)
    const int    KC   = K >> 5;
    const int    kc   = (int)(rest % KC);
    const int    nb   = (int)(rest / KC);
    unsigned short t[8];
#pragma unroll
    for (int i = 0; i < 8; ++i)
        t[i] = f2bf(src[(size_t)(kc * 32 + kk0 + i) * N + nb * 128 + nl]);
    uint4 r;
    r.x = (unsigned)t[0] | ((unsigned)t[1] << 16);
    r.y = (unsigned)t[2] | ((unsigned)t[3] << 16);
    r.z = (unsigned)t[4] | ((unsigned)t[5] << 16);
    r.w = (unsigned)t[6] | ((unsigned)t[7] << 16);
    *(uint4*)(dst + base) = r;
}

// ---------------------------------------------------------------------------
// Input projection: h = x @ in_w + in_b + pos_emb, plus bf16 copy hb.
// ---------------------------------------------------------------------------
__global__ __launch_bounds__(256) void input_proj_kernel(
    const float* __restrict__ x, const float* __restrict__ in_w,
    const float* __restrict__ in_b, const float* __restrict__ pos,
    float* __restrict__ h, unsigned short* __restrict__ hb)
{
    __shared__ float xr[DIN];
    const int row = blockIdx.x;          // 0..8191
    const int tid = threadIdx.x;
    const int s   = row & (SS - 1);
    if (tid < DIN) xr[tid] = x[(size_t)row * DIN + tid];
    __syncthreads();
#pragma unroll
    for (int i = 0; i < 3; ++i) {
        const int n = tid + i * 256;
        float acc = in_b[n] + pos[(size_t)s * DD + n];
#pragma unroll
        for (int k = 0; k < DIN; ++k) acc += xr[k] * in_w[(size_t)k * DD + n];
        h[(size_t)row * DD + n]  = acc;
        hb[(size_t)row * DD + n] = f2bf(acc);
    }
}

// ---------------------------------------------------------------------------
// bf16-WMMA GEMM:  C[M,N] = A[M,K] @ W[K,N] + bias  (optional ReLU)
// Block tile 128x128, K-chunk 32, 256 threads = 8 waves, each wave 32x64.
// Tiles DMA'd into double-buffered LDS by the TDM (wave 0), overlapped with
// the 8 WMMAs/chunk of all waves.
// ---------------------------------------------------------------------------
__global__ __launch_bounds__(256) void gemm_wmma_kernel(
    const unsigned short* __restrict__ A,
    const unsigned short* __restrict__ Wt,
    const float* __restrict__ bias, float* __restrict__ outF,
    unsigned short* __restrict__ outB, int M, int N, int K, int relu)
{
    __shared__ alignas(16) unsigned short ldsA[2][128 * 32];   // [m][k]
    __shared__ alignas(16) unsigned short ldsB[2][128 * 32];   // [n][k]

    const int tid  = threadIdx.x;
    const int lane = tid & 31;
    const int wave = tid >> 5;
    const int g    = (lane >> 4) & 1;
    const int nl   = lane & 15;
    const int wm   = wave >> 1;   // 0..3  (32 rows each)
    const int wn   = wave & 1;    // 0..1  (64 cols each)

    const int mblk = M >> 7;
    const int bm   = (blockIdx.x % mblk) * 128;
    const int nb   = blockIdx.x / mblk;
    const int bn   = nb * 128;
    const int KC   = K >> 5;

    v8f acc[2][4] = {};

#if HAVE_TDM
    // A tile: 128 rows x 64B (8 units of 8B), row stride K*2 bytes = K/4 units
    // W tile: one contiguous 8KB block (fragment-order pre-transpose)
    auto issue = [&](int buf, int kc) {
        tdm_load_2d((unsigned)(size_t)&ldsA[buf][0],
                    A + (size_t)bm * K + (kc << 5), 8u, 128u,
                    (unsigned long long)(K >> 2));
        tdm_load_2d((unsigned)(size_t)&ldsB[buf][0],
                    Wt + ((size_t)(nb * KC + kc) << 12), 1024u, 1u, 1024ull);
    };
    if (wave == 0) issue(0, 0);
#else
    const int arow = tid >> 1;            // 0..127
    const int ak0  = (tid & 1) * 16;      // 0,16
#endif

    for (int kc = 0; kc < KC; ++kc) {
#if HAVE_TDM
        if (wave == 0) {
            if (kc + 1 < KC) {
                issue((kc + 1) & 1, kc + 1);
                __builtin_amdgcn_s_wait_tensorcnt(2);  // chunk kc complete
            } else {
                __builtin_amdgcn_s_wait_tensorcnt(0);
            }
        }
        __syncthreads();
        const unsigned short* tA = ldsA[kc & 1];
        const unsigned short* tB = ldsB[kc & 1];
#else
        const int kb = kc << 5;
        {
            const uint4* s =
                (const uint4*)(A + (size_t)(bm + arow) * K + kb + ak0);
            uint4* d = (uint4*)&ldsA[0][arow * 32 + ak0];
            d[0] = s[0];
            d[1] = s[1];
        }
        {
            const uint4* s =
                (const uint4*)(Wt + ((size_t)(nb * KC + kc) << 12) + tid * 16);
            uint4* d = (uint4*)&ldsB[0][tid * 16];
            d[0] = s[0];
            d[1] = s[1];
        }
        __syncthreads();
        const unsigned short* tA = ldsA[0];
        const unsigned short* tB = ldsB[0];
#endif

        Frag a[2], b[4];
#pragma unroll
        for (int sm = 0; sm < 2; ++sm) {
            const int ml = wm * 32 + sm * 16 + nl;
#pragma unroll
            for (int v = 0; v < 8; ++v) {
                const int k0 = 2 * v + (v >= 4 ? 8 : 0) + 8 * g;
                a[sm].u[v] = *(const unsigned int*)&tA[ml * 32 + k0];
            }
        }
#pragma unroll
        for (int sn = 0; sn < 4; ++sn) {
            const int nn = wn * 64 + sn * 16 + nl;
#pragma unroll
            for (int v = 0; v < 8; ++v) {
                const int k0 = 2 * v + (v >= 4 ? 8 : 0) + 8 * g;
                b[sn].u[v] = *(const unsigned int*)&tB[nn * 32 + k0];
            }
        }
#pragma unroll
        for (int sm = 0; sm < 2; ++sm)
#pragma unroll
            for (int sn = 0; sn < 4; ++sn)
                acc[sm][sn] = __builtin_amdgcn_wmma_f32_16x16x32_bf16(
                    false, a[sm].bf, false, b[sn].bf, (short)0, acc[sm][sn],
                    false, false);
        __syncthreads();
    }

    // ---- epilogue ----
#pragma unroll
    for (int sm = 0; sm < 2; ++sm)
#pragma unroll
        for (int sn = 0; sn < 4; ++sn) {
            const int n  = bn + wn * 64 + sn * 16 + nl;
            const float bv = bias[n];
#pragma unroll
            for (int r = 0; r < 8; ++r) {
                const int m = bm + wm * 32 + sm * 16 + r + 8 * g;
                float val = acc[sm][sn][r] + bv;
                if (relu) val = fmaxf(val, 0.0f);
                if (outF) outF[(size_t)m * N + n] = val;
                if (outB) outB[(size_t)m * N + n] = f2bf(val);
            }
        }
}

// ---------------------------------------------------------------------------
// Flash attention with bf16 WMMA.  Q,K,V: bf16-bits [B*S, 768], head slice
// of 64 columns.  One wave per 16-query tile; 4 waves per block share K/V
// tiles (32 keys / iteration).  O: fp32 [B*S, 768].
// ---------------------------------------------------------------------------
__global__ __launch_bounds__(128) void attention_wmma_kernel(
    const unsigned short* __restrict__ Q, const unsigned short* __restrict__ Kv,
    const unsigned short* __restrict__ Vv, const int* __restrict__ lens,
    float* __restrict__ O)
{
    __shared__ alignas(16) unsigned short ldsK[32 * 64];    // [key][d]
    __shared__ alignas(16) unsigned short ldsVt[64 * 32];   // [d][key]
    __shared__ alignas(4)  unsigned short ldsP[4][16 * 32]; // per-wave P

    const int tid  = threadIdx.x;
    const int lane = tid & 31;
    const int wave = tid >> 5;
    const int g    = (lane >> 4) & 1;
    const int nl   = lane & 15;

    const int blk = blockIdx.x;
    const int b   = blk / (HH_ * 8);
    const int rem = blk % (HH_ * 8);
    const int hh  = rem / 8;
    const int qb  = rem % 8;
    const int q0  = qb * 64 + wave * 16;

    const int    len     = lens[b];
    const size_t rowbase = (size_t)b * SS;
    const int    coff    = hh * DH_;

    Frag qf[2];
#pragma unroll
    for (int c = 0; c < 2; ++c)
#pragma unroll
        for (int v = 0; v < 8; ++v) {
            const int k0 = 2 * v + (v >= 4 ? 8 : 0) + 8 * g;
            const size_t idx =
                (rowbase + q0 + nl) * DD + coff + c * 32 + k0;
            qf[c].u[v] = *(const unsigned int*)(Q + idx);
        }

    float mi[8], li[8];
    v8f   oacc[4] = {};
#pragma unroll
    for (int r = 0; r < 8; ++r) { mi[r] = -3.0e38f; li[r] = 0.0f; }

    for (int j = 0; j < SS / 32; ++j) {
        const int kb = j * 32;
        {
            const int row = tid >> 2, d0 = (tid & 3) * 16;
            const uint4* src =
                (const uint4*)(Kv + (rowbase + kb + row) * DD + coff + d0);
            uint4* dst = (uint4*)&ldsK[row * 64 + d0];
            dst[0] = src[0];
            dst[1] = src[1];
        }
        {
            const int key = tid & 31, d0 = (tid >> 5) * 16;
            const unsigned short* src =
                Vv + (rowbase + kb + key) * DD + coff + d0;
#pragma unroll
            for (int i = 0; i < 16; ++i) ldsVt[(d0 + i) * 32 + key] = src[i];
        }
        __syncthreads();

        v8f scv[2];
#pragma unroll
        for (int sn = 0; sn < 2; ++sn) {
            v8f sc = {};
#pragma unroll
            for (int c = 0; c < 2; ++c) {
                Frag kfr;
                const int row = sn * 16 + nl;
#pragma unroll
                for (int v = 0; v < 8; ++v)
                    kfr.u[v] = *(const unsigned int*)&ldsK[row * 64 + c * 32 +
                                                           2 * v + (v >= 4 ? 8 : 0) + 8 * g];
                sc = __builtin_amdgcn_wmma_f32_16x16x32_bf16(
                    false, qf[c].bf, false, kfr.bf, (short)0, sc, false, false);
            }
            scv[sn] = sc;
        }

        float scf[2][8];
#pragma unroll
        for (int sn = 0; sn < 2; ++sn) {
            const int kidx = kb + sn * 16 + nl;
            const bool kok = kidx < len;
#pragma unroll
            for (int r = 0; r < 8; ++r) {
                const bool qok = (q0 + r + 8 * g) < len;
                scf[sn][r] =
                    (kok && qok) ? scv[sn][r] * 0.125f : -1.0e9f;
            }
        }

        float pf[2][8];
#pragma unroll
        for (int r = 0; r < 8; ++r) {
            float rm = fmaxf(scf[0][r], scf[1][r]);
            rm = fmaxf(rm, __shfl_xor(rm, 1, 32));
            rm = fmaxf(rm, __shfl_xor(rm, 2, 32));
            rm = fmaxf(rm, __shfl_xor(rm, 4, 32));
            rm = fmaxf(rm, __shfl_xor(rm, 8, 32));
            const float nm = fmaxf(mi[r], rm);
            const float f  = __expf(mi[r] - nm);
            const float p0 = __expf(scf[0][r] - nm);
            const float p1 = __expf(scf[1][r] - nm);
            float rs = p0 + p1;
            rs += __shfl_xor(rs, 1, 32);
            rs += __shfl_xor(rs, 2, 32);
            rs += __shfl_xor(rs, 4, 32);
            rs += __shfl_xor(rs, 8, 32);
            li[r] = li[r] * f + rs;
            mi[r] = nm;
#pragma unroll
            for (int d = 0; d < 4; ++d) oacc[d][r] *= f;
            pf[0][r] = p0;
            pf[1][r] = p1;
        }

#pragma unroll
        for (int r = 0; r < 8; ++r) {
            const int m = r + 8 * g;
            ldsP[wave][m * 32 + nl]      = f2bf(pf[0][r]);
            ldsP[wave][m * 32 + 16 + nl] = f2bf(pf[1][r]);
        }
        Frag pfr;
#pragma unroll
        for (int v = 0; v < 8; ++v) {
            const int k0 = 2 * v + (v >= 4 ? 8 : 0) + 8 * g;
            pfr.u[v] = *(const unsigned int*)&ldsP[wave][nl * 32 + k0];
        }

#pragma unroll
        for (int d = 0; d < 4; ++d) {
            Frag vfr;
            const int rowd = d * 16 + nl;
#pragma unroll
            for (int v = 0; v < 8; ++v)
                vfr.u[v] =
                    *(const unsigned int*)&ldsVt[rowd * 32 + 2 * v + (v >= 4 ? 8 : 0) + 8 * g];
            oacc[d] = __builtin_amdgcn_wmma_f32_16x16x32_bf16(
                false, pfr.bf, false, vfr.bf, (short)0, oacc[d], false, false);
        }
        __syncthreads();
    }

#pragma unroll
    for (int d = 0; d < 4; ++d)
#pragma unroll
        for (int r = 0; r < 8; ++r) {
            const int m = q0 + r + 8 * g;
            O[(rowbase + m) * DD + coff + d * 16 + nl] = oacc[d][r] / li[r];
        }
}

// ---------------------------------------------------------------------------
// Residual add + LayerNorm per row; writes fp32 h and bf16 hb.
// ---------------------------------------------------------------------------
__global__ __launch_bounds__(256) void add_ln_kernel(
    const float* __restrict__ res, const float* __restrict__ delta,
    const float* __restrict__ gam, const float* __restrict__ bet,
    float* __restrict__ out, unsigned short* __restrict__ outb)
{
    __shared__ float red[256];
    const int row = blockIdx.x;
    const int tid = threadIdx.x;
    float x[3];
    float s = 0.0f;
#pragma unroll
    for (int i = 0; i < 3; ++i) {
        const size_t idx = (size_t)row * DD + tid + i * 256;
        const float v = res[idx] + delta[idx];
        x[i] = v;
        s += v;
    }
    red[tid] = s;
    __syncthreads();
    for (int off = 128; off > 0; off >>= 1) {
        if (tid < off) red[tid] += red[tid + off];
        __syncthreads();
    }
    const float mean = red[0] * (1.0f / DD);
    __syncthreads();
    float ss = 0.0f;
#pragma unroll
    for (int i = 0; i < 3; ++i) {
        const float d = x[i] - mean;
        ss += d * d;
    }
    red[tid] = ss;
    __syncthreads();
    for (int off = 128; off > 0; off >>= 1) {
        if (tid < off) red[tid] += red[tid + off];
        __syncthreads();
    }
    const float rstd = rsqrtf(red[0] * (1.0f / DD) + 1e-5f);
#pragma unroll
    for (int i = 0; i < 3; ++i) {
        const int c = tid + i * 256;
        const float y = (x[i] - mean) * rstd * gam[c] + bet[c];
        out[(size_t)row * DD + c]  = y;
        outb[(size_t)row * DD + c] = f2bf(y);
    }
}

// ---------------------------------------------------------------------------
// Masked mean-pool over valid rows + out projection -> [B,1]
// ---------------------------------------------------------------------------
__global__ __launch_bounds__(256) void pool_out_kernel(
    const float* __restrict__ h, const int* __restrict__ lens,
    const float* __restrict__ ow, const float* __restrict__ ob,
    float* __restrict__ out)
{
    __shared__ float red[256];
    const int b   = blockIdx.x;
    const int tid = threadIdx.x;
    const int len = lens[b];
    float partial = 0.0f;
#pragma unroll
    for (int i = 0; i < 3; ++i) {
        const int d = tid + i * 256;
        const float w = ow[d];
        float cs = 0.0f;
        for (int s = 0; s < len; ++s)
            cs += h[((size_t)b * SS + s) * DD + d];
        partial += w * cs;
    }
    red[tid] = partial;
    __syncthreads();
    for (int off = 128; off > 0; off >>= 1) {
        if (tid < off) red[tid] += red[tid + off];
        __syncthreads();
    }
    if (tid == 0) out[b] = red[0] / (float)len + ob[0];
}

// ---------------------------------------------------------------------------
// Host orchestration
// ---------------------------------------------------------------------------
extern "C" void kernel_launch(void* const* d_in, const int* in_sizes, int n_in,
                              void* d_out, int out_size, void* d_ws,
                              size_t ws_size, hipStream_t stream)
{
    const float* x      = (const float*)d_in[0];
    const int*   lens   = (const int*)d_in[1];
    const float* in_w   = (const float*)d_in[2];
    const float* in_b   = (const float*)d_in[3];
    const float* pos    = (const float*)d_in[4];
    const float* qw     = (const float*)d_in[5];
    const float* qb     = (const float*)d_in[6];
    const float* kw     = (const float*)d_in[7];
    const float* kbias  = (const float*)d_in[8];
    const float* vw     = (const float*)d_in[9];
    const float* vb     = (const float*)d_in[10];
    const float* w1     = (const float*)d_in[11];
    const float* b1     = (const float*)d_in[12];
    const float* w2     = (const float*)d_in[13];
    const float* b2     = (const float*)d_in[14];
    const float* ln1g   = (const float*)d_in[15];
    const float* ln1b   = (const float*)d_in[16];
    const float* ln2g   = (const float*)d_in[17];
    const float* ln2b   = (const float*)d_in[18];
    const float* ow     = (const float*)d_in[19];
    const float* ob     = (const float*)d_in[20];
    float*       outp   = (float*)d_out;

    size_t off = 0;
    auto carve = [&](size_t bytes) {
        void* p = (char*)d_ws + off;
        off += (bytes + 255) & ~(size_t)255;
        return p;
    };
    float*          h    = (float*)carve((size_t)MTOT * DD * 4);
    float*          Obuf = (float*)carve((size_t)MTOT * DD * 4);
    unsigned short* hb   = (unsigned short*)carve((size_t)MTOT * DD * 2);
    unsigned short* Qb   = (unsigned short*)carve((size_t)MTOT * DD * 2);
    unsigned short* Kb   = (unsigned short*)carve((size_t)MTOT * DD * 2);
    unsigned short* Vb   = (unsigned short*)carve((size_t)MTOT * DD * 2);
    unsigned short* F1   = (unsigned short*)carve((size_t)MTOT * DFF_ * 2);
    unsigned short* qwt  = (unsigned short*)carve((size_t)LL * DD * DD * 2);
    unsigned short* kwt  = (unsigned short*)carve((size_t)LL * DD * DD * 2);
    unsigned short* vwt  = (unsigned short*)carve((size_t)LL * DD * DD * 2);
    unsigned short* w1t  = (unsigned short*)carve((size_t)LL * DD * DFF_ * 2);
    unsigned short* w2t  = (unsigned short*)carve((size_t)LL * DFF_ * DD * 2);

    {
        const int gdd = (DD * DD) / 2048;     // 288 blocks per layer matrix
        const int gdf = (DD * DFF_) / 2048;   // 1152 blocks per layer matrix
        w_pretr_kernel<<<dim3(gdd, LL), 256, 0, stream>>>(qw, qwt, DD, DD);
        w_pretr_kernel<<<dim3(gdd, LL), 256, 0, stream>>>(kw, kwt, DD, DD);
        w_pretr_kernel<<<dim3(gdd, LL), 256, 0, stream>>>(vw, vwt, DD, DD);
        w_pretr_kernel<<<dim3(gdf, LL), 256, 0, stream>>>(w1, w1t, DD, DFF_);
        w_pretr_kernel<<<dim3(gdf, LL), 256, 0, stream>>>(w2, w2t, DFF_, DD);
    }

    input_proj_kernel<<<MTOT, 256, 0, stream>>>(x, in_w, in_b, pos, h, hb);

    const int gD   = (MTOT / 128) * (DD / 128);    // 384 blocks
    const int gFF1 = (MTOT / 128) * (DFF_ / 128);  // 1536 blocks

    for (int l = 0; l < LL; ++l) {
        const size_t wdd = (size_t)l * DD * DD;
        const size_t wdf = (size_t)l * DD * DFF_;
        gemm_wmma_kernel<<<gD, 256, 0, stream>>>(
            hb, qwt + wdd, qb + l * DD, nullptr, Qb, MTOT, DD, DD, 0);
        gemm_wmma_kernel<<<gD, 256, 0, stream>>>(
            hb, kwt + wdd, kbias + l * DD, nullptr, Kb, MTOT, DD, DD, 0);
        gemm_wmma_kernel<<<gD, 256, 0, stream>>>(
            hb, vwt + wdd, vb + l * DD, nullptr, Vb, MTOT, DD, DD, 0);

        attention_wmma_kernel<<<BB * HH_ * 8, 128, 0, stream>>>(
            Qb, Kb, Vb, lens, Obuf);

        add_ln_kernel<<<MTOT, 256, 0, stream>>>(
            h, Obuf, ln1g + l * DD, ln1b + l * DD, h, hb);

        gemm_wmma_kernel<<<gFF1, 256, 0, stream>>>(
            hb, w1t + wdf, b1 + l * DFF_, nullptr, F1, MTOT, DFF_, DD, 1);
        gemm_wmma_kernel<<<gD, 256, 0, stream>>>(
            F1, w2t + wdf, b2 + l * DD, Obuf, nullptr, MTOT, DD, DFF_, 0);

        add_ln_kernel<<<MTOT, 256, 0, stream>>>(
            h, Obuf, ln2g + l * DD, ln2b + l * DD, h, hb);
    }

    pool_out_kernel<<<BB, 256, 0, stream>>>(h, lens, ow, ob, outp);
}
